// Cylinder3D_27178553049619
// MI455X (gfx1250) — matrix operand order repaired and checked by reading, not compiled, not run on hardware
//
#include <hip/hip_runtime.h>

typedef __attribute__((ext_vector_type(16))) _Float16 v16h;
typedef __attribute__((ext_vector_type(8)))  float    v8f;

#define PPW 2   // waves per block in pass kernels (2 waves x 24KB LDS + 2KB = 50KB)
#define FPW 4   // waves per block in final kernel

// LDS sub-tile strides (halves)
#define TSA 2048   // 16 x 128
#define TSB 4096   // 16 x 256

union FragU { uint4 q[2]; v16h h; };
union Half4 { _Float16 h[4]; uint2 u; };

// A fragment (16x32 f16, M x K) from a row-major [16][strideH] LDS tile.
// ISA 7.12.2: lane<16 holds row m=lane, K = {kt*32+0..7, kt*32+16..23};
// lane>=16 holds K = {kt*32+8..15, kt*32+24..31}. Two 16B ds loads per lane.
__device__ __forceinline__ v16h frag_a_lds(const _Float16* base, int strideH, int kt, int lane) {
  const int m = lane & 15, g = lane >> 4;
  const uint4* p = (const uint4*)(base + m * strideH + kt * 32 + g * 8);
  FragU u; u.q[0] = p[0]; u.q[1] = p[2];   // +16 halves = +2 uint4
  return u.h;
}

// B fragment (32x16 f16, K x N) from pre-swizzled weights: 512 halves per
// fragment, lane-major (lane's 16 halves contiguous -> two 16B global loads).
__device__ __forceinline__ v16h frag_b_glb(const _Float16* wf, int frag, int lane) {
  const uint4* p = (const uint4*)(wf + ((size_t)frag << 9)) + (lane << 1);
  FragU u; u.q[0] = p[0]; u.q[1] = p[1];
  return u.h;
}

// order-preserving float <-> uint for atomicMax-based segment_max
__device__ __forceinline__ unsigned enc_ord(float f) {
  unsigned b = __float_as_uint(f);
  return (b & 0x80000000u) ? ~b : (b | 0x80000000u);
}
__device__ __forceinline__ float dec_ord(unsigned u) {
  unsigned b = (u & 0x80000000u) ? (u & 0x7fffffffu) : ~u;
  float f = __uint_as_float(b);
  return __builtin_isfinite(f) ? f : 0.0f;   // init value 0 decodes to NaN -> 0
}

// hidden layer, M-blocked: each B fragment is loaded once and applied to MB
// A tiles (MB accumulator sets) -> halves global weight-fragment traffic.
template <int MB>
__device__ __forceinline__ void layer_fwd(const _Float16* hin, int strideIn, int tsIn, int KT,
                                          const _Float16* wf, int NT,
                                          const float* scale, const float* shift,
                                          _Float16* hout, int strideOut, int tsOut, int lane) {
  const int n = lane & 15, g = lane >> 4;
  for (int nt = 0; nt < NT; ++nt) {
    v8f c[MB] = {};
    for (int kt = 0; kt < KT; ++kt) {
      v16h b = frag_b_glb(wf, kt * NT + nt, lane);
#pragma unroll
      for (int mi = 0; mi < MB; ++mi) {
        v16h a = frag_a_lds(hin + mi * tsIn, strideIn, kt, lane);
        c[mi] = __builtin_amdgcn_wmma_f32_16x16x32_f16(false, a, false, b, (short)0, c[mi], false, false);
      }
    }
    const float sc = scale[nt * 16 + n];
    const float sh = shift[nt * 16 + n];
#pragma unroll
    for (int mi = 0; mi < MB; ++mi) {
#pragma unroll
      for (int v = 0; v < 8; ++v) {
        float val = fmaxf(c[mi][v] * sc + sh, 0.0f);
        hout[mi * tsOut + (v + 8 * g) * strideOut + nt * 16 + n] = (_Float16)val;
      }
    }
  }
}

// terminal layer of a stats pass: per-feature sum/sumsq of (matmul + bias)
template <int MB>
__device__ __forceinline__ void layer_stats(const _Float16* hin, int strideIn, int tsIn, int KT,
                                            const _Float16* wf, int NT,
                                            const float* bias,
                                            float* sSum, float* sQ, int lane) {
  const int n = lane & 15;
  for (int nt = 0; nt < NT; ++nt) {
    v8f c[MB] = {};
    for (int kt = 0; kt < KT; ++kt) {
      v16h b = frag_b_glb(wf, kt * NT + nt, lane);
#pragma unroll
      for (int mi = 0; mi < MB; ++mi) {
        v16h a = frag_a_lds(hin + mi * tsIn, strideIn, kt, lane);
        c[mi] = __builtin_amdgcn_wmma_f32_16x16x32_f16(false, a, false, b, (short)0, c[mi], false, false);
      }
    }
    const float bb = bias[nt * 16 + n];
    float s = 0.f, q = 0.f;
#pragma unroll
    for (int mi = 0; mi < MB; ++mi) {
#pragma unroll
      for (int v = 0; v < 8; ++v) { float val = c[mi][v] + bb; s += val; q += val * val; }
    }
    atomicAdd(&sSum[nt * 16 + n], s);
    atomicAdd(&sQ[nt * 16 + n], q);
  }
}

// terminal layer of pass 4: matmul + bias, scatter-max into pooled[V][256]
template <int MB>
__device__ __forceinline__ void layer_scatter(const _Float16* hin, int strideIn, int tsIn, int KT,
                                              const _Float16* wf, int NT,
                                              const float* bias, const int* vix,
                                              unsigned* pooled, int lane) {
  const int n = lane & 15;
  for (int nt = 0; nt < NT; ++nt) {
    v8f c[MB] = {};
    for (int kt = 0; kt < KT; ++kt) {
      v16h b = frag_b_glb(wf, kt * NT + nt, lane);
#pragma unroll
      for (int mi = 0; mi < MB; ++mi) {
        v16h a = frag_a_lds(hin + mi * tsIn, strideIn, kt, lane);
        c[mi] = __builtin_amdgcn_wmma_f32_16x16x32_f16(false, a, false, b, (short)0, c[mi], false, false);
      }
    }
    const float bb = bias[nt * 16 + n];
#pragma unroll
    for (int mi = 0; mi < MB; ++mi) {
#pragma unroll
      for (int v = 0; v < 8; ++v) {
        float val = c[mi][v] + bb;
        atomicMax(&pooled[(size_t)vix[mi * 8 + v] * 256 + nt * 16 + n], enc_ord(val));
      }
    }
  }
}

// BN0(x) -> A[mi][16][32] f16 (K padded 16 -> 32 with zeros), vectorized
template <int MB>
__device__ __forceinline__ void fill_h0(const float* __restrict__ x, int base,
                                        const float* __restrict__ sc0,
                                        const float* __restrict__ sh0,
                                        _Float16* A, int lane) {
  const uint2 zz = {0u, 0u};
  for (int e = lane; e < 64 * MB; e += 32) {
    const int mi = e >> 6;
    const int r = e & 63;
    const int m = r >> 2, fq = r & 3;
    const float4 xv = ((const float4*)x)[(size_t)(base + mi * 16 + m) * 4 + fq];
    const float4 sv = ((const float4*)sc0)[fq];
    const float4 tv = ((const float4*)sh0)[fq];
    Half4 o;
    o.h[0] = (_Float16)(xv.x * sv.x + tv.x);
    o.h[1] = (_Float16)(xv.y * sv.y + tv.y);
    o.h[2] = (_Float16)(xv.z * sv.z + tv.z);
    o.h[3] = (_Float16)(xv.w * sv.w + tv.w);
    *(uint2*)(A + mi * TSA + m * 128 + fq * 4) = o.u;
    *(uint2*)(A + mi * TSA + m * 128 + 16 + fq * 4) = zz;
  }
}

// recompute chain through layer PASS (stats) or full chain + scatter (PASS 4)
template <int PASS, int MB>
__device__ __forceinline__ void pp_chain(const int* __restrict__ unq, int base,
                                         _Float16* A, _Float16* B,
                                         const _Float16* w1f, const float* sc1, const float* sh1,
                                         const _Float16* w2f, const float* sc2, const float* sh2,
                                         const _Float16* w3f, const float* sc3, const float* sh3,
                                         const _Float16* w4f, const float* biasT,
                                         float* sSum, float* sQ, unsigned* pooled, int lane) {
  if (PASS == 1) { layer_stats<MB>(A, 128, TSA, 1, w1f, 4, biasT, sSum, sQ, lane); return; }
  layer_fwd<MB>(A, 128, TSA, 1, w1f, 4, sc1, sh1, B, 256, TSB, lane);            // 16 -> 64
  if (PASS == 2) { layer_stats<MB>(B, 256, TSB, 2, w2f, 8, biasT, sSum, sQ, lane); return; }
  layer_fwd<MB>(B, 256, TSB, 2, w2f, 8, sc2, sh2, A, 128, TSA, lane);            // 64 -> 128
  if (PASS == 3) { layer_stats<MB>(A, 128, TSA, 4, w3f, 16, biasT, sSum, sQ, lane); return; }
  layer_fwd<MB>(A, 128, TSA, 4, w3f, 16, sc3, sh3, B, 256, TSB, lane);           // 128 -> 256
  int vix[MB * 8];
  const int g = lane >> 4;
#pragma unroll
  for (int mi = 0; mi < MB; ++mi)
#pragma unroll
    for (int v = 0; v < 8; ++v) vix[mi * 8 + v] = unq[base + mi * 16 + v + 8 * g];
  layer_scatter<MB>(B, 256, TSB, 8, w4f, 16, biasT, vix, pooled, lane);          // 256 -> 256 + max
}

template <int PASS>
__global__ __launch_bounds__(PPW * 32)
void pp_pass(const float* __restrict__ x, const int* __restrict__ unq, int ntiles,
             const float* __restrict__ sc0, const float* __restrict__ sh0,
             const _Float16* __restrict__ w1f,
             const float* __restrict__ sc1, const float* __restrict__ sh1,
             const _Float16* __restrict__ w2f,
             const float* __restrict__ sc2, const float* __restrict__ sh2,
             const _Float16* __restrict__ w3f,
             const float* __restrict__ sc3, const float* __restrict__ sh3,
             const _Float16* __restrict__ w4f,
             const float* __restrict__ biasT,
             float* __restrict__ gS, float* __restrict__ gQ,
             unsigned* __restrict__ pooled) {
  __shared__ _Float16 bufA[PPW][2 * TSA];   // h0 (32 wide), h2 (128 wide), 2 M-tiles
  __shared__ _Float16 bufB[PPW][2 * TSB];   // h1 (64 wide), h3 (256 wide), 2 M-tiles
  __shared__ float sSum[256], sQ[256];
  const int lane = threadIdx.x & 31;
  const int wv   = threadIdx.x >> 5;
  if (PASS < 4) {
    for (int i = threadIdx.x; i < 256; i += blockDim.x) { sSum[i] = 0.f; sQ[i] = 0.f; }
    __syncthreads();
  }
  const int wgl = blockIdx.x * PPW + wv;
  const int nw  = gridDim.x * PPW;
  _Float16* A = bufA[wv];
  _Float16* B = bufB[wv];
  const int npairs = (ntiles + 1) >> 1;
  for (int tp = wgl; tp < npairs; tp += nw) {
    const int t0 = tp * 2;
    const int base = t0 * 16;
    if (t0 + 1 < ntiles) {            // wave-uniform branch: EXEC stays full
      fill_h0<2>(x, base, sc0, sh0, A, lane);
      pp_chain<PASS, 2>(unq, base, A, B, w1f, sc1, sh1, w2f, sc2, sh2,
                        w3f, sc3, sh3, w4f, biasT, sSum, sQ, pooled, lane);
    } else {
      fill_h0<1>(x, base, sc0, sh0, A, lane);
      pp_chain<PASS, 1>(unq, base, A, B, w1f, sc1, sh1, w2f, sc2, sh2,
                        w3f, sc3, sh3, w4f, biasT, sSum, sQ, pooled, lane);
    }
  }
  if (PASS < 4) {
    __syncthreads();
    const int F = (PASS == 1) ? 64 : (PASS == 2) ? 128 : 256;
    for (int f = threadIdx.x; f < F; f += blockDim.x) {
      atomicAdd(&gS[f], sSum[f]);
      atomicAdd(&gQ[f], sQ[f]);
    }
  }
}

// zero pooled buffer (ordered-uint -inf) and stats accumulators each call
__global__ void pp_init(unsigned* __restrict__ pooled, size_t npool,
                        float* __restrict__ stats, int nstats) {
  const size_t i = (size_t)blockIdx.x * blockDim.x + threadIdx.x;
  const size_t stride = (size_t)gridDim.x * blockDim.x;
  for (size_t k = i; k < npool; k += stride) pooled[k] = 0u;
  if (i < (size_t)nstats) stats[i] = 0.f;
}

// pre-swizzle f32 [K][Nout] weights into WMMA B fragments (f16, lane-major)
__global__ void pp_swizzle(const float* __restrict__ W, int Kreal, int KT, int NT,
                           _Float16* __restrict__ out) {
  const int nout  = NT * 16;
  const int total = KT * NT * 512;
  for (int i = blockIdx.x * blockDim.x + threadIdx.x; i < total; i += gridDim.x * blockDim.x) {
    const int frag = i >> 9, r = i & 511;
    const int l = r >> 4, e = r & 15;
    const int kt = frag / NT, nt = frag - kt * NT;
    const int n = nt * 16 + (l & 15);
    const int k = kt * 32 + (e >> 3) * 16 + (l >> 4) * 8 + (e & 7);
    const float v = (k < Kreal) ? W[(size_t)k * nout + n] : 0.0f;
    out[i] = (_Float16)v;
  }
}

// per-feature mean/var of the raw input (16 features)
__global__ void pp_stats0(const float* __restrict__ x, int N,
                          float* __restrict__ gS, float* __restrict__ gQ) {
  __shared__ float sS[16], sQm[16];
  if (threadIdx.x < 16) { sS[threadIdx.x] = 0.f; sQm[threadIdx.x] = 0.f; }
  __syncthreads();
  const int f  = threadIdx.x & 15;
  const int p0 = (blockIdx.x * blockDim.x + threadIdx.x) >> 4;
  const int ps = (gridDim.x * blockDim.x) >> 4;
  float s = 0.f, q = 0.f;
  for (int p = p0; p < N; p += ps) {
    const float v = x[(size_t)p * 16 + f];
    s += v; q += v * v;
  }
  atomicAdd(&sS[f], s); atomicAdd(&sQm[f], q);
  __syncthreads();
  if (threadIdx.x < 16) {
    atomicAdd(&gS[threadIdx.x], sS[threadIdx.x]);
    atomicAdd(&gQ[threadIdx.x], sQm[threadIdx.x]);
  }
}

// fold bias + BN into per-feature scale/shift: relu((C+b)*s + t) = relu(C*s + (b*s+t))
__global__ void pp_finalize(const float* __restrict__ gS, const float* __restrict__ gQ,
                            const float* __restrict__ gamma, const float* __restrict__ beta,
                            const float* __restrict__ bias, float* __restrict__ scale,
                            float* __restrict__ shift, int F, float invN) {
  const int f = blockIdx.x * blockDim.x + threadIdx.x;
  if (f >= F) return;
  const float mean = gS[f] * invN;
  const float var  = gQ[f] * invN - mean * mean;
  const float s = gamma[f] * rsqrtf(var + 1e-5f);
  const float t = beta[f] - mean * s;
  const float b = bias ? bias[f] : 0.0f;
  scale[f] = s;
  shift[f] = b * s + t;
}

// pooled[V][256] -> relu(pooled @ wc + bc) -> out[V][16]
__global__ __launch_bounds__(FPW * 32)
void pp_final(const unsigned* __restrict__ pooled, const _Float16* __restrict__ wcf,
              const float* __restrict__ bc, float* __restrict__ out, int V) {
  __shared__ _Float16 buf[FPW][16 * 256];
  const int lane = threadIdx.x & 31;
  const int wv = threadIdx.x >> 5;
  const int n = lane & 15, g = lane >> 4;
  const int wgl = blockIdx.x * FPW + wv;
  const int nw = gridDim.x * FPW;
  const int nfull  = V >> 4;
  const int ntiles = (V + 15) >> 4;
  _Float16* B = buf[wv];
  for (int tile = wgl; tile < ntiles; tile += nw) {
    const int base = tile * 16;
    if (tile < nfull) {
      // full tile: vectorized decode fill, no guards anywhere
      for (int e = lane; e < 1024; e += 32) {
        const int m = e >> 6, q = e & 63;
        const uint4 pv = ((const uint4*)pooled)[((size_t)(base + m) << 6) + q];
        Half4 o;
        o.h[0] = (_Float16)dec_ord(pv.x);
        o.h[1] = (_Float16)dec_ord(pv.y);
        o.h[2] = (_Float16)dec_ord(pv.z);
        o.h[3] = (_Float16)dec_ord(pv.w);
        *(uint2*)(B + m * 256 + q * 4) = o.u;
      }
    } else {
      // tail tile (V % 16 rows): guarded scalar fill
      for (int e = lane; e < 16 * 256; e += 32) {
        const int m = e >> 8, f = e & 255;
        const int vx = base + m;
        float val = 0.f;
        if (vx < V) val = dec_ord(pooled[(size_t)vx * 256 + f]);
        B[m * 256 + f] = (_Float16)val;
      }
    }
    v8f c = {};
    for (int kt = 0; kt < 8; ++kt) {
      v16h a = frag_a_lds(B, 256, kt, lane);
      v16h b = frag_b_glb(wcf, kt, lane);
      c = __builtin_amdgcn_wmma_f32_16x16x32_f16(false, a, false, b, (short)0, c, false, false);
    }
    const float bb = bc[n];
    if (tile < nfull) {
#pragma unroll
      for (int v = 0; v < 8; ++v)
        out[(size_t)(base + v + 8 * g) * 16 + n] = fmaxf(c[v] + bb, 0.0f);
    } else {
#pragma unroll
      for (int v = 0; v < 8; ++v) {
        const int vx = base + v + 8 * g;
        if (vx < V) out[(size_t)vx * 16 + n] = fmaxf(c[v] + bb, 0.0f);
      }
    }
  }
}

extern "C" void kernel_launch(void* const* d_in, const int* in_sizes, int n_in,
                              void* d_out, int out_size, void* d_ws, size_t ws_size,
                              hipStream_t stream) {
  (void)n_in; (void)ws_size;
  const float* x    = (const float*)d_in[0];
  const float* bn0g = (const float*)d_in[1];
  const float* bn0b = (const float*)d_in[2];
  const float* w1   = (const float*)d_in[3];
  const float* b1   = (const float*)d_in[4];
  const float* bn1g = (const float*)d_in[5];
  const float* bn1b = (const float*)d_in[6];
  const float* w2   = (const float*)d_in[7];
  const float* b2   = (const float*)d_in[8];
  const float* bn2g = (const float*)d_in[9];
  const float* bn2b = (const float*)d_in[10];
  const float* w3   = (const float*)d_in[11];
  const float* b3   = (const float*)d_in[12];
  const float* bn3g = (const float*)d_in[13];
  const float* bn3b = (const float*)d_in[14];
  const float* w4   = (const float*)d_in[15];
  const float* b4   = (const float*)d_in[16];
  const float* wc   = (const float*)d_in[17];
  const float* bc   = (const float*)d_in[18];
  const int*   unq  = (const int*)d_in[19];
  float* out = (float*)d_out;

  const int N = in_sizes[0] / 16;   // 600000
  const int V = out_size / 16;      // 200000 (num_voxels lives on device; derive from out)
  const int ntiles = N / 16;

  // ---- workspace carve-up ----
  char* ws = (char*)d_ws;
  size_t off = 0;
  auto take = [&](size_t bytes) -> char* {
    char* p = ws + off;
    off = (off + bytes + 255) & ~(size_t)255;
    return p;
  };
  unsigned* pooled   = (unsigned*)take((size_t)V * 256 * 4);   // ~204.8 MB
  float* statsBase   = (float*)take(928 * 4);
  float* gS0 = statsBase +   0; float* gQ0 = statsBase +  16;
  float* gS1 = statsBase +  32; float* gQ1 = statsBase +  96;
  float* gS2 = statsBase + 160; float* gQ2 = statsBase + 288;
  float* gS3 = statsBase + 416; float* gQ3 = statsBase + 672;
  float* scBase = (float*)take(928 * 4);
  float* sc0 = scBase +   0; float* sh0 = scBase +  16;
  float* sc1 = scBase +  32; float* sh1 = scBase +  96;
  float* sc2 = scBase + 160; float* sh2 = scBase + 288;
  float* sc3 = scBase + 416; float* sh3 = scBase + 672;
  _Float16* w1f = (_Float16*)take( 2048 * 2);   // 1 kt  x  4 nt
  _Float16* w2f = (_Float16*)take( 8192 * 2);   // 2 kt  x  8 nt
  _Float16* w3f = (_Float16*)take(32768 * 2);   // 4 kt  x 16 nt
  _Float16* w4f = (_Float16*)take(65536 * 2);   // 8 kt  x 16 nt
  _Float16* wcf = (_Float16*)take( 4096 * 2);   // 8 kt  x  1 nt

  const float invN = 1.0f / (float)N;

  // 0) re-init pooled + stats every call (deterministic under graph replay)
  pp_init<<<2048, 256, 0, stream>>>(pooled, (size_t)V * 256, statsBase, 928);

  // 1) pre-swizzle weights into WMMA B-fragment layout
  pp_swizzle<<<8,   256, 0, stream>>>(w1,  16, 1,  4, w1f);
  pp_swizzle<<<32,  256, 0, stream>>>(w2,  64, 2,  8, w2f);
  pp_swizzle<<<128, 256, 0, stream>>>(w3, 128, 4, 16, w3f);
  pp_swizzle<<<256, 256, 0, stream>>>(w4, 256, 8, 16, w4f);
  pp_swizzle<<<16,  256, 0, stream>>>(wc, 256, 8,  1, wcf);

  // 2) BN0 stats -> scale/shift
  pp_stats0<<<1024, 256, 0, stream>>>(x, N, gS0, gQ0);
  pp_finalize<<<1, 16, 0, stream>>>(gS0, gQ0, bn0g, bn0b, nullptr, sc0, sh0, 16, invN);

  // 3) stats passes for BN1..BN3 (WMMA recompute chains, 2x M-blocked)
  pp_pass<1><<<2048, PPW * 32, 0, stream>>>(x, unq, ntiles, sc0, sh0, w1f, sc1, sh1, w2f,
                                            sc2, sh2, w3f, sc3, sh3, w4f, b1, gS1, gQ1, pooled);
  pp_finalize<<<1, 64, 0, stream>>>(gS1, gQ1, bn1g, bn1b, b1, sc1, sh1, 64, invN);

  pp_pass<2><<<2048, PPW * 32, 0, stream>>>(x, unq, ntiles, sc0, sh0, w1f, sc1, sh1, w2f,
                                            sc2, sh2, w3f, sc3, sh3, w4f, b2, gS2, gQ2, pooled);
  pp_finalize<<<1, 128, 0, stream>>>(gS2, gQ2, bn2g, bn2b, b2, sc2, sh2, 128, invN);

  pp_pass<3><<<2048, PPW * 32, 0, stream>>>(x, unq, ntiles, sc0, sh0, w1f, sc1, sh1, w2f,
                                            sc2, sh2, w3f, sc3, sh3, w4f, b3, gS3, gQ3, pooled);
  pp_finalize<<<1, 256, 0, stream>>>(gS3, gQ3, bn3g, bn3b, b3, sc3, sh3, 256, invN);

  // 4) full chain + segment-max scatter
  pp_pass<4><<<2048, PPW * 32, 0, stream>>>(x, unq, ntiles, sc0, sh0, w1f, sc1, sh1, w2f,
                                            sc2, sh2, w3f, sc3, sh3, w4f, b4, nullptr, nullptr, pooled);

  // 5) voxel compression: relu(pooled @ wc + bc)
  pp_final<<<1024, FPW * 32, 0, stream>>>(pooled, wcf, bc, out, V);
}
